// MultiHeadLatentAttention_81707457839331
// MI455X (gfx1250) — compile-verified, matrix operand order checked by reading, not run on hardware
//
#include <hip/hip_runtime.h>
#include <cstdint>
#include <cstddef>

// Problem constants (match reference)
#define BB   2
#define SS   2048
#define HIDD 2048
#define NHH  16
#define NOPE_D 128
#define ROPE_D 64
#define VD_D 128
#define QKD_D 192
#define KVR_D 512
#define QR_D 1536
#define MS   (BB * SS)          // 4096 token rows
#define ATT_SCALE 0.0721687836487032f   // 192^-0.5

typedef __bf16 bf16_t;
typedef __attribute__((ext_vector_type(16))) __bf16 v16bf;
typedef __attribute__((ext_vector_type(8)))  __bf16 v8bf;
typedef __attribute__((ext_vector_type(8)))  float  v8f;

__device__ __forceinline__ v8f wmma_bf16(v16bf a, v16bf b, v8f c) {
  return __builtin_amdgcn_wmma_f32_16x16x32_bf16(false, a, false, b, (short)0, c,
                                                 false, false);
}

// Load a 16x32 bf16 WMMA fragment from a row-major matrix where `row..row+15`
// indexes the M (or N) dimension and the contraction dim is contiguous.
// Per-lane layout (16-bit A/B matrix, wave32):
//   lanes 0-15  : row = lane,    K = {k0+0..7,  k0+16..23}
//   lanes 16-31 : row = lane-16, K = {k0+8..15, k0+24..31}
__device__ __forceinline__ v16bf load_frag(const bf16_t* p, int ld, int row, int k0) {
  const int lane = threadIdx.x & 31;
  const int r    = lane & 15;
  const int half = lane >> 4;
  const bf16_t* base = p + (size_t)(row + r) * ld + k0 + half * 8;
  union { v16bf v; v8bf h[2]; } u;
  u.h[0] = *(const v8bf*)(base);        // 16B -> b128 load
  u.h[1] = *(const v8bf*)(base + 16);   // 16B -> b128 load
  return u.v;
}

// Store a 16x16 f32 C/D tile: lane holds col n = lane&15, rows (lane>>4)*8 + i.
__device__ __forceinline__ void store_tile(float* C, int ldc, int row, int col, v8f acc) {
  const int lane = threadIdx.x & 31;
  const int r    = lane & 15;
  const int half = lane >> 4;
#pragma unroll
  for (int i = 0; i < 8; ++i)
    C[(size_t)(row + half * 8 + i) * ldc + col + r] = acc[i];
}

// ---------------------------------------------------------------------------
// Generic NT GEMM: C[M,N] = A[M,K] * W[N,K]^T  (bf16 in, f32 out)
// 4 waves / block; each wave owns a 32(M) x 64(N) tile:
//   2 A-fragments x 4 B-fragments -> 8 WMMAs per 12 b128 loads per k-step.
// k-loop manually unrolled 2x with ping-pong fragment register sets (aA/bA vs
// aB/bB) so the pipeline needs no register rotation movs and each WMMA's
// s_wait_loadcnt covers loads issued a full 32-K-step earlier.
// Requires M%32==0, N%64==0, K%64==0 — all satisfied here.
// ---------------------------------------------------------------------------
__global__ void __launch_bounds__(128)
gemm_nt_kernel(const bf16_t* __restrict__ A, const bf16_t* __restrict__ W,
               float* __restrict__ C, int M, int N, int K) {
  const int wid  = threadIdx.x >> 5;
  const int wg   = blockIdx.x * 4 + wid;
  const int tn64 = N >> 6;
  const int tm   = wg / tn64;
  const int tn   = wg % tn64;
  const int row0 = tm * 32;
  const int col0 = tn * 64;
  if (row0 >= M) return;

  v8f acc[2][4] = {};
  v16bf aA[2], bA[4], aB[2], bB[4];
  aA[0] = load_frag(A, K, row0,      0);
  aA[1] = load_frag(A, K, row0 + 16, 0);
#pragma unroll
  for (int j = 0; j < 4; ++j) bA[j] = load_frag(W, K, col0 + j * 16, 0);

  for (int k0 = 0; k0 < K; k0 += 64) {
    // refill ping buffers for k0+32 (always in range: K%64==0, k0 <= K-64)
    __builtin_prefetch(A + (size_t)row0 * K + k0 + 64, 0, 1);
    aB[0] = load_frag(A, K, row0,      k0 + 32);
    aB[1] = load_frag(A, K, row0 + 16, k0 + 32);
#pragma unroll
    for (int j = 0; j < 4; ++j) bB[j] = load_frag(W, K, col0 + j * 16, k0 + 32);
    // compute on even-stage fragments (loaded one stage earlier)
#pragma unroll
    for (int j = 0; j < 4; ++j) {
      acc[0][j] = wmma_bf16(aA[0], bA[j], acc[0][j]);
      acc[1][j] = wmma_bf16(aA[1], bA[j], acc[1][j]);
    }
    // refill pong buffers for k0+64 (guarded tail)
    if (k0 + 64 < K) {
      aA[0] = load_frag(A, K, row0,      k0 + 64);
      aA[1] = load_frag(A, K, row0 + 16, k0 + 64);
#pragma unroll
      for (int j = 0; j < 4; ++j) bA[j] = load_frag(W, K, col0 + j * 16, k0 + 64);
    }
    // compute on odd-stage fragments
#pragma unroll
    for (int j = 0; j < 4; ++j) {
      acc[0][j] = wmma_bf16(aB[0], bB[j], acc[0][j]);
      acc[1][j] = wmma_bf16(aB[1], bB[j], acc[1][j]);
    }
  }
#pragma unroll
  for (int j = 0; j < 4; ++j) {
    store_tile(C, N, row0,      col0 + j * 16, acc[0][j]);
    store_tile(C, N, row0 + 16, col0 + j * 16, acc[1][j]);
  }
}

// ---------------------------------------------------------------------------
// fp32 -> bf16 cast
// ---------------------------------------------------------------------------
__global__ void __launch_bounds__(256)
cast_kernel(const float* __restrict__ x, bf16_t* __restrict__ y, size_t n) {
  size_t i = (size_t)blockIdx.x * 256 + threadIdx.x;
  if (i < n) y[i] = (bf16_t)x[i];
}

// ---------------------------------------------------------------------------
// RMSNorm over row of length R, multiply by gamma, emit bf16.
// One 256-thread block per row.
// ---------------------------------------------------------------------------
__global__ void __launch_bounds__(256)
rmsnorm_cast_kernel(const float* __restrict__ X, const float* __restrict__ gamma,
                    bf16_t* __restrict__ Y, int R, float invR) {
  const int row = blockIdx.x;
  const float* x = X + (size_t)row * R;
  float ss = 0.f;
  for (int i = threadIdx.x; i < R; i += 256) { float v = x[i]; ss += v * v; }
#pragma unroll
  for (int off = 16; off > 0; off >>= 1) ss += __shfl_xor(ss, off, 32);
  __shared__ float red[8];
  __shared__ float sinv;
  if ((threadIdx.x & 31) == 0) red[threadIdx.x >> 5] = ss;
  __syncthreads();
  if (threadIdx.x == 0) {
    float t = 0.f;
    for (int i = 0; i < 8; ++i) t += red[i];
    sinv = rsqrtf(t * invR + 1e-6f);
  }
  __syncthreads();
  const float inv = sinv;
  bf16_t* y = Y + (size_t)row * R;
  for (int i = threadIdx.x; i < R; i += 256)
    y[i] = (bf16_t)(x[i] * inv * gamma[i]);
}

// ---------------------------------------------------------------------------
// KV branch: RMSNorm first 512 of kv row -> bf16 c_kv; RoPE last 64 -> bf16.
// One 256-thread block per (b,s) row.
// ---------------------------------------------------------------------------
__global__ void __launch_bounds__(256)
kv_norm_rope_kernel(const float* __restrict__ KV, const float* __restrict__ gamma,
                    const float* __restrict__ cosp, const float* __restrict__ sinp,
                    bf16_t* __restrict__ CKV, bf16_t* __restrict__ KR) {
  const int row = blockIdx.x;  // b*S + s
  const float* x = KV + (size_t)row * (KVR_D + ROPE_D);
  float ss = 0.f;
  for (int i = threadIdx.x; i < KVR_D; i += 256) { float v = x[i]; ss += v * v; }
#pragma unroll
  for (int off = 16; off > 0; off >>= 1) ss += __shfl_xor(ss, off, 32);
  __shared__ float red[8];
  __shared__ float sinv;
  if ((threadIdx.x & 31) == 0) red[threadIdx.x >> 5] = ss;
  __syncthreads();
  if (threadIdx.x == 0) {
    float t = 0.f;
    for (int i = 0; i < 8; ++i) t += red[i];
    sinv = rsqrtf(t * (1.0f / KVR_D) + 1e-6f);
  }
  __syncthreads();
  const float inv = sinv;
  for (int i = threadIdx.x; i < KVR_D; i += 256)
    CKV[(size_t)row * KVR_D + i] = (bf16_t)(x[i] * inv * gamma[i]);
  if (threadIdx.x < ROPE_D) {
    const int d = threadIdx.x;
    float v   = x[KVR_D + d];
    float rot = (d < 32) ? -x[KVR_D + d + 32] : x[KVR_D + d - 32];
    float c = cosp[(size_t)row * ROPE_D + d];
    float s = sinp[(size_t)row * ROPE_D + d];
    KR[(size_t)row * ROPE_D + d] = (bf16_t)(v * c + rot * s);
  }
}

// ---------------------------------------------------------------------------
// Q: apply RoPE to last 64 dims of each head, repack into [b, h, s, 192] bf16.
// One 192-thread block per (b,s,h).
// ---------------------------------------------------------------------------
__global__ void __launch_bounds__(192)
q_rope_pack_kernel(const float* __restrict__ Qf, const float* __restrict__ cosp,
                   const float* __restrict__ sinp, bf16_t* __restrict__ Qb) {
  const int blk = blockIdx.x;
  const int h  = blk & (NHH - 1);
  const int bs = blk >> 4;                 // b*S + s
  const int d  = threadIdx.x;              // 0..191
  const float* src = Qf + (size_t)bs * (NHH * QKD_D) + h * QKD_D;
  float val;
  if (d < NOPE_D) {
    val = src[d];
  } else {
    const int dd = d - NOPE_D;
    float v   = src[d];
    float rot = (dd < 32) ? -src[NOPE_D + dd + 32] : src[NOPE_D + dd - 32];
    val = v * cosp[(size_t)bs * ROPE_D + dd] + rot * sinp[(size_t)bs * ROPE_D + dd];
  }
  const int b = bs >> 11, s = bs & (SS - 1);
  Qb[((size_t)(b * NHH + h) * SS + s) * QKD_D + d] = (bf16_t)val;
}

// ---------------------------------------------------------------------------
// kvx split: k_nope -> K[b,h,s,0:128], broadcast k_rope -> K[b,h,s,128:192],
// v -> V^T[b,h,d,s] (pre-transposed so P*V B-fragments are K-contiguous).
// One 128-thread block per (b,s,h).
// ---------------------------------------------------------------------------
__global__ void __launch_bounds__(128)
kvx_split_kernel(const float* __restrict__ KVX, const bf16_t* __restrict__ KR,
                 bf16_t* __restrict__ Kb, bf16_t* __restrict__ Vt) {
  const int blk = blockIdx.x;
  const int h  = blk & (NHH - 1);
  const int bs = blk >> 4;
  const int d  = threadIdx.x;              // 0..127
  const float* src = KVX + (size_t)bs * (NHH * (NOPE_D + VD_D)) + h * (NOPE_D + VD_D);
  const int b = bs >> 11, s = bs & (SS - 1);
  const size_t bhs = (size_t)(b * NHH + h) * SS + s;
  Kb[bhs * QKD_D + d] = (bf16_t)src[d];
  Vt[((size_t)(b * NHH + h) * VD_D + d) * SS + s] = (bf16_t)src[NOPE_D + d];
  if (d < ROPE_D)
    Kb[bhs * QKD_D + NOPE_D + d] = KR[(size_t)bs * ROPE_D + d];
}

// ---------------------------------------------------------------------------
// Causal flash attention, one wave per 16-query tile, 32-key blocks.
// Q,K: [b,h,s,192] bf16; Vt: [b,h,128,s] bf16; O: [b,s,h*128] bf16.
// Block = 4 independent waves (no block-wide sync: trip counts differ).
// Per 32-key block: 12 WMMAs Q*K^T + 8 WMMAs P*V.
// ---------------------------------------------------------------------------
__global__ void __launch_bounds__(128)
mla_attn_kernel(const bf16_t* __restrict__ Q, const bf16_t* __restrict__ Km,
                const bf16_t* __restrict__ Vt, bf16_t* __restrict__ O) {
  __shared__ bf16_t pbuf[4][16][40];   // per-wave 16x32 P tile, 16B-aligned rows
  const int wid  = threadIdx.x >> 5;
  const int lane = threadIdx.x & 31;
  const int r    = lane & 15;
  const int half = lane >> 4;
  const int bh  = blockIdx.x >> 5;     // (b*NH + h)
  const int grp = blockIdx.x & 31;
  const int q0  = (grp * 4 + wid) * 16;

  const bf16_t* Qp = Q  + (size_t)bh * SS * QKD_D;
  const bf16_t* Kp = Km + (size_t)bh * SS * QKD_D;
  const bf16_t* Vp = Vt + (size_t)bh * VD_D * SS;

  v16bf qf[6];
#pragma unroll
  for (int t = 0; t < 6; ++t) qf[t] = load_frag(Qp, QKD_D, q0, t * 32);

  v8f o[8] = {};
  float mrow[8], lrow[8];
#pragma unroll
  for (int i = 0; i < 8; ++i) { mrow[i] = -1e30f; lrow[i] = 0.f; }

  const int nkb = (q0 + 15) / 32 + 1;  // causal bound: keys <= q0+15
  for (int kb = 0; kb < nkb; ++kb) {
    const int kbase = kb * 32;
    // --- scores: two 16x16 tiles over 192-dim contraction ---
    v8f s0 = {}, s1 = {};
#pragma unroll
    for (int t = 0; t < 6; ++t) {
      v16bf b0 = load_frag(Kp, QKD_D, kbase,      t * 32);
      v16bf b1 = load_frag(Kp, QKD_D, kbase + 16, t * 32);
      s0 = wmma_bf16(qf[t], b0, s0);
      s1 = wmma_bf16(qf[t], b1, s1);
    }
    // --- scale + causal mask; element (m = half*8+i, n = r) ---
    float rm[8];
#pragma unroll
    for (int i = 0; i < 8; ++i) {
      const int qg = q0 + half * 8 + i;
      s0[i] = (kbase + r      <= qg) ? s0[i] * ATT_SCALE : -1e30f;
      s1[i] = (kbase + 16 + r <= qg) ? s1[i] * ATT_SCALE : -1e30f;
      rm[i] = fmaxf(s0[i], s1[i]);
    }
    // row max across the 16 lanes holding each row (stays within half)
#pragma unroll
    for (int off = 1; off < 16; off <<= 1)
#pragma unroll
      for (int i = 0; i < 8; ++i)
        rm[i] = fmaxf(rm[i], __shfl_xor(rm[i], off, 32));
    // --- online softmax update ---
    float rs[8];
#pragma unroll
    for (int i = 0; i < 8; ++i) {
      const float mn = fmaxf(mrow[i], rm[i]);
      const float al = __expf(mrow[i] - mn);
      mrow[i] = mn;
      s0[i] = __expf(s0[i] - mn);
      s1[i] = __expf(s1[i] - mn);
      rs[i] = s0[i] + s1[i];
      lrow[i] *= al;
#pragma unroll
      for (int t = 0; t < 8; ++t) o[t][i] *= al;
    }
#pragma unroll
    for (int off = 1; off < 16; off <<= 1)
#pragma unroll
      for (int i = 0; i < 8; ++i)
        rs[i] += __shfl_xor(rs[i], off, 32);
#pragma unroll
    for (int i = 0; i < 8; ++i) lrow[i] += rs[i];
    // --- C-layout P -> LDS -> A-fragment layout ---
#pragma unroll
    for (int i = 0; i < 8; ++i) {
      pbuf[wid][half * 8 + i][r]      = (bf16_t)s0[i];
      pbuf[wid][half * 8 + i][16 + r] = (bf16_t)s1[i];
    }
    asm volatile("s_wait_dscnt 0" ::: "memory");  // wave-local LDS RAW fence
    v16bf pa = load_frag(&pbuf[wid][0][0], 40, 0, 0);
    // --- P(16x32) * V(32x128) via 8 WMMAs on Vt rows ---
#pragma unroll
    for (int t = 0; t < 8; ++t) {
      v16bf vb = load_frag(Vp, SS, t * 16, kbase);
      o[t] = wmma_bf16(pa, vb, o[t]);
    }
  }
  // --- normalize and store to [b, s, h*128] ---
  const int b = bh >> 4, h = bh & (NHH - 1);
  bf16_t* Op = O + (size_t)b * SS * (NHH * VD_D) + (size_t)h * VD_D;
#pragma unroll
  for (int i = 0; i < 8; ++i) {
    const float inv = 1.f / lrow[i];
    const int qrow  = q0 + half * 8 + i;
#pragma unroll
    for (int t = 0; t < 8; ++t)
      Op[(size_t)qrow * (NHH * VD_D) + t * 16 + r] = (bf16_t)(o[t][i] * inv);
  }
}

// ---------------------------------------------------------------------------
// Host-side orchestration
// ---------------------------------------------------------------------------
extern "C" void kernel_launch(void* const* d_in, const int* in_sizes, int n_in,
                              void* d_out, int out_size, void* d_ws, size_t ws_size,
                              hipStream_t stream) {
  const float* hidden   = (const float*)d_in[0];
  const float* cosp     = (const float*)d_in[1];
  const float* sinp     = (const float*)d_in[2];
  const float* Wq_down  = (const float*)d_in[3];
  const float* q_gamma  = (const float*)d_in[4];
  const float* Wq_up    = (const float*)d_in[5];
  const float* Wkv_down = (const float*)d_in[6];
  const float* kv_gamma = (const float*)d_in[7];
  const float* Wkv_up   = (const float*)d_in[8];
  const float* Wo       = (const float*)d_in[9];
  float* out = (float*)d_out;

  char* ws = (char*)d_ws;
  size_t off = 0;
  auto take = [&](size_t bytes) -> char* {
    char* p = ws + off;
    off = (off + bytes + 255) & ~(size_t)255;
    return p;
  };

  // bf16 buffers
  bf16_t* hid_bf   = (bf16_t*)take((size_t)MS * HIDD * 2);
  bf16_t* wqd_bf   = (bf16_t*)take((size_t)QR_D * HIDD * 2);
  bf16_t* wqu_bf   = (bf16_t*)take((size_t)NHH * QKD_D * QR_D * 2);
  bf16_t* wkvd_bf  = (bf16_t*)take((size_t)(KVR_D + ROPE_D) * HIDD * 2);
  bf16_t* wkvu_bf  = (bf16_t*)take((size_t)NHH * (NOPE_D + VD_D) * KVR_D * 2);
  bf16_t* wo_bf    = (bf16_t*)take((size_t)HIDD * NHH * VD_D * 2);
  bf16_t* qlat_bf  = (bf16_t*)take((size_t)MS * QR_D * 2);
  bf16_t* Qbf      = (bf16_t*)take((size_t)BB * NHH * SS * QKD_D * 2);
  bf16_t* ckv_bf   = (bf16_t*)take((size_t)MS * KVR_D * 2);
  bf16_t* krope_bf = (bf16_t*)take((size_t)MS * ROPE_D * 2);
  bf16_t* Kbf      = (bf16_t*)take((size_t)BB * NHH * SS * QKD_D * 2);
  bf16_t* Vt_bf    = (bf16_t*)take((size_t)BB * NHH * VD_D * SS * 2);
  bf16_t* attn_bf  = (bf16_t*)take((size_t)MS * NHH * VD_D * 2);
  // shared f32 scratch (largest transient: kvx 4096x4096)
  float*  scratch  = (float*)take((size_t)MS * 4096 * 4);
  (void)ws_size; (void)n_in; (void)in_sizes; (void)out_size;

  auto cast = [&](const float* src, bf16_t* dst, size_t n) {
    cast_kernel<<<(unsigned)((n + 255) / 256), 256, 0, stream>>>(src, dst, n);
  };
  auto gemm = [&](const bf16_t* A, const bf16_t* W, float* C, int M, int N, int K) {
    const int blocks = (M / 32) * (N / 64) / 4;   // 4 waves (tiles) per block
    gemm_nt_kernel<<<blocks, 128, 0, stream>>>(A, W, C, M, N, K);
  };

  // 0) casts to bf16
  cast(hidden,   hid_bf,  (size_t)MS * HIDD);
  cast(Wq_down,  wqd_bf,  (size_t)QR_D * HIDD);
  cast(Wq_up,    wqu_bf,  (size_t)NHH * QKD_D * QR_D);
  cast(Wkv_down, wkvd_bf, (size_t)(KVR_D + ROPE_D) * HIDD);
  cast(Wkv_up,   wkvu_bf, (size_t)NHH * (NOPE_D + VD_D) * KVR_D);
  cast(Wo,       wo_bf,   (size_t)HIDD * NHH * VD_D);

  // 1) q_lat = rmsnorm(hidden @ Wq_down^T)
  gemm(hid_bf, wqd_bf, scratch, MS, QR_D, HIDD);
  rmsnorm_cast_kernel<<<MS, 256, 0, stream>>>(scratch, q_gamma, qlat_bf,
                                              QR_D, 1.0f / QR_D);
  // 2) q = q_lat @ Wq_up^T, then RoPE + head-major pack
  gemm(qlat_bf, wqu_bf, scratch, MS, NHH * QKD_D, QR_D);
  q_rope_pack_kernel<<<MS * NHH, 192, 0, stream>>>(scratch, cosp, sinp, Qbf);

  // 3) kv = hidden @ Wkv_down^T; split/norm/rope
  gemm(hid_bf, wkvd_bf, scratch, MS, KVR_D + ROPE_D, HIDD);
  kv_norm_rope_kernel<<<MS, 256, 0, stream>>>(scratch, kv_gamma, cosp, sinp,
                                              ckv_bf, krope_bf);
  // 4) kvx = c_kv @ Wkv_up^T; split into K (nope+rope) and V^T
  gemm(ckv_bf, wkvu_bf, scratch, MS, NHH * (NOPE_D + VD_D), KVR_D);
  kvx_split_kernel<<<MS * NHH, 128, 0, stream>>>(scratch, krope_bf, Kbf, Vt_bf);

  // 5) causal flash attention: 32 (b,h) pairs x 32 blocks of 4 query tiles
  mla_attn_kernel<<<BB * NHH * 32, 128, 0, stream>>>(Qbf, Kbf, Vt_bf, attn_bf);

  // 6) out = attn @ Wo^T  (f32 to d_out)
  gemm(attn_bf, wo_bf, out, MS, HIDD, NHH * VD_D);
}
